// BasisCustAttention_23888608100429
// MI455X (gfx1250) — matrix operand order repaired and checked by reading, not compiled, not run on hardware
//
#include <hip/hip_runtime.h>
#include <cstdint>
#include <cstddef>

#define BB 32
#define SS 2048
#define DD 512
#define NEGV (-1e9f)

typedef __attribute__((ext_vector_type(16))) __bf16          v16bf;
typedef __attribute__((ext_vector_type(8)))  float           v8f;
typedef __attribute__((ext_vector_type(8)))  unsigned short  us8;
typedef __attribute__((ext_vector_type(16))) unsigned short  us16;

__device__ __forceinline__ unsigned short f2bf(float f) {
    unsigned u = __float_as_uint(f);
    u += 0x7FFFu + ((u >> 16) & 1u);   // round-to-nearest-even
    return (unsigned short)(u >> 16);
}

// ---------------------------------------------------------------------------
// We [K=512][N=512] fp32  ->  WeT [N=512][K=512] bf16 (as ushort bits)
// ---------------------------------------------------------------------------
__global__ __launch_bounds__(256)
void k_weT(const float* __restrict__ We, unsigned short* __restrict__ WeT) {
    int idx = blockIdx.x * 256 + threadIdx.x;       // 0 .. 262143
    int n = idx >> 9, k = idx & 511;
    WeT[idx] = f2bf(We[k * DD + n]);
}

// ---------------------------------------------------------------------------
// q[d]  = b3[d] + (1/S) * sum_nb W3[nb][d]        (exact softmax collapse)
// c[d'] = battn[d'] + sum_d q[d] * Wq[d][d']
// ---------------------------------------------------------------------------
__global__ __launch_bounds__(512)
void k_cvec(const float* __restrict__ W3, const float* __restrict__ b3,
            const float* __restrict__ Wq, const float* __restrict__ battn,
            float* __restrict__ cvec) {
    __shared__ float q[DD];
    int t = threadIdx.x;
    float s = 0.f;
#pragma unroll
    for (int nb = 0; nb < 8; ++nb) s += W3[nb * DD + t];
    q[t] = b3[t] + s * (1.0f / (float)SS);
    __syncthreads();
    float acc = battn[t];
    for (int k = 0; k < DD; ++k) acc += q[k] * Wq[k * DD + t];
    cvec[t] = acc;
}

// ---------------------------------------------------------------------------
// Pass A: scores[r] = sum_d v[d] * tanh( (x @ We)[r,d] + c[d] )
// One block = 64 rows x all 512 N.  8 waves, each wave owns an N-chunk of 64
// and iterates 4 M-subtiles so each B fragment is reused 4x from registers.
// bf16 WMMA 16x16x32, fp32 accumulate, K = 512 in 16 steps.
// ---------------------------------------------------------------------------
__global__ __launch_bounds__(256)
void k_scores(const float* __restrict__ x, const unsigned short* __restrict__ WeT,
              const float* __restrict__ cvec, const float* __restrict__ vvec,
              float* __restrict__ scores) {
    __shared__ unsigned short Atile[64][32];   // 64 rows x 32 K, bf16 bits (4 KB)
    __shared__ float scRed[64];

    const int tid  = threadIdx.x;
    const int lane = tid & 31;
    const int wv   = tid >> 5;                 // wave id 0..7
    const long row0 = (long)blockIdx.x * 64;   // global row (b*S + s)
    const int n0   = wv * 64;                  // this wave's N base

    for (int i = tid; i < 64; i += 256) scRed[i] = 0.f;

    v8f acc[4][4] = {};                        // [m-subtile][n-tile] 16x16 fp32

    const int mrow  = lane & 15;               // row-in-tile / B column
    const int khalf = lane >> 4;               // lane-group K half
    // staging coords: 2048 elems / 256 threads = 8 contiguous floats each
    const int sm = (tid * 8) >> 5;             // staged row 0..63
    const int sc = (tid * 8) & 31;             // staged col {0,8,16,24}

    for (int kk = 0; kk < DD; kk += 32) {
        // ---- stage A tile: 64x32 fp32 -> bf16 -> LDS (2 x b128 per thread) ----
        {
            const float* src = x + (row0 + sm) * DD + kk + sc;
            float4 f0 = *(const float4*)src;
            float4 f1 = *(const float4*)(src + 4);
            us8 pk = { f2bf(f0.x), f2bf(f0.y), f2bf(f0.z), f2bf(f0.w),
                       f2bf(f1.x), f2bf(f1.y), f2bf(f1.z), f2bf(f1.w) };
            *(us8*)&Atile[sm][sc] = pk;
            if (kk + 32 < DD)
                __builtin_prefetch(src + 32, 0, 3);   // near-cache prefetch of next K slice
        }
        __syncthreads();

        // ---- B fragments for this wave's 4 N-tiles (held in registers) ----
        const int kb = kk + khalf * 8;
        v16bf bfr[4];
#pragma unroll
        for (int t = 0; t < 4; ++t) {
            const unsigned short* bp = WeT + (size_t)(n0 + t * 16 + mrow) * DD + kb;
            us8  b0 = *(const us8*)bp;
            us8  b1 = *(const us8*)(bp + 16);
            us16 bu = __builtin_shufflevector(b0, b1, 0, 1, 2, 3, 4, 5, 6, 7,
                                              8, 9, 10, 11, 12, 13, 14, 15);
            bfr[t] = __builtin_bit_cast(v16bf, bu);
        }

        // ---- 4 M-subtiles x 4 N-tiles of WMMA, B reused from registers ----
#pragma unroll
        for (int ms = 0; ms < 4; ++ms) {
            const int ar = ms * 16 + mrow;
            us8  a0 = *(const us8*)&Atile[ar][khalf * 8];
            us8  a1 = *(const us8*)&Atile[ar][16 + khalf * 8];
            us16 au = __builtin_shufflevector(a0, a1, 0, 1, 2, 3, 4, 5, 6, 7,
                                              8, 9, 10, 11, 12, 13, 14, 15);
            v16bf af = __builtin_bit_cast(v16bf, au);
#pragma unroll
            for (int t = 0; t < 4; ++t)
                acc[ms][t] = __builtin_amdgcn_wmma_f32_16x16x32_bf16(
                                 false, af, false, bfr[t], (short)0, acc[ms][t],
                                 false, false);
        }
        __syncthreads();
    }

    // ---- fused epilogue: sum_d v[d]*tanh(y+c[d]) per row ----
    float cv[4], vw[4];
#pragma unroll
    for (int t = 0; t < 4; ++t) {
        int d = n0 + t * 16 + mrow;            // this lane's N column (C/D layout)
        cv[t] = cvec[d];
        vw[t] = vvec[d];
    }
#pragma unroll
    for (int ms = 0; ms < 4; ++ms) {
        float rowp[8] = {0.f, 0.f, 0.f, 0.f, 0.f, 0.f, 0.f, 0.f};
#pragma unroll
        for (int t = 0; t < 4; ++t)
#pragma unroll
            for (int j = 0; j < 8; ++j)        // VGPR j -> row M = j + 8*khalf
                rowp[j] += vw[t] * tanhf(acc[ms][t][j] + cv[t]);
        // butterfly over the 16 lanes sharing a row (N reduction in-wave)
#pragma unroll
        for (int off = 1; off < 16; off <<= 1)
#pragma unroll
            for (int j = 0; j < 8; ++j)
                rowp[j] += __shfl_xor(rowp[j], off, 32);
        if (mrow == 0) {                       // lanes 0 and 16 carry the sums
            const int rb = ms * 16 + khalf * 8;
#pragma unroll
            for (int j = 0; j < 8; ++j)
                atomicAdd(&scRed[rb + j], rowp[j]);   // ds_add_f32, 8-wave combine
        }
    }
    __syncthreads();
    if (tid < 64) scores[row0 + tid] = scRed[tid];
}

// ---------------------------------------------------------------------------
// Pass B1: per-batch masked max + sum(exp); zero output row.
// ---------------------------------------------------------------------------
__global__ __launch_bounds__(256)
void k_maxsum(const unsigned char* __restrict__ mask, float* __restrict__ scores,
              float* __restrict__ maxv, float* __restrict__ sumv,
              float* __restrict__ out) {
    __shared__ float red[256];
    const int b = blockIdx.x, tid = threadIdx.x;
    float lm = -3.4e38f;
    for (int s = tid; s < SS; s += 256) {
        float sc = scores[b * SS + s];
        if (!mask[b * SS + s]) sc = NEGV;
        scores[b * SS + s] = sc;               // store masked score for pass B2
        lm = fmaxf(lm, sc);
    }
    red[tid] = lm; __syncthreads();
    for (int off = 128; off > 0; off >>= 1) {
        if (tid < off) red[tid] = fmaxf(red[tid], red[tid + off]);
        __syncthreads();
    }
    const float mx = red[0]; __syncthreads();
    float ls = 0.f;
    for (int s = tid; s < SS; s += 256) ls += expf(scores[b * SS + s] - mx);
    red[tid] = ls; __syncthreads();
    for (int off = 128; off > 0; off >>= 1) {
        if (tid < off) red[tid] += red[tid + off];
        __syncthreads();
    }
    if (tid == 0) { maxv[b] = mx; sumv[b] = red[0]; }
    for (int d = tid; d < DD; d += 256) out[b * DD + d] = 0.f;
}

// ---------------------------------------------------------------------------
// Pass B2: out[b,d] += sum_s alpha[s] * x[b,s,d]  (8 s-chunks per batch)
// ---------------------------------------------------------------------------
__global__ __launch_bounds__(512)
void k_wsum(const float* __restrict__ x, const float* __restrict__ scores,
            const float* __restrict__ maxv, const float* __restrict__ sumv,
            float* __restrict__ out) {
    const int b = blockIdx.x >> 3, chunk = blockIdx.x & 7;
    const int d = threadIdx.x;
    const float mx = maxv[b], inv = 1.0f / sumv[b];
    const int s0 = chunk * 256;
    const float* xp = x + ((size_t)b * SS + s0) * DD + d;
    const float* sp = scores + b * SS + s0;
    float acc = 0.f;
    for (int i = 0; i < 256; ++i)
        acc += expf(sp[i] - mx) * xp[(size_t)i * DD];
    atomicAdd(&out[b * DD + d], acc * inv);
}

// ---------------------------------------------------------------------------
extern "C" void kernel_launch(void* const* d_in, const int* in_sizes, int n_in,
                              void* d_out, int out_size, void* d_ws, size_t ws_size,
                              hipStream_t stream) {
    const float*         x     = (const float*)d_in[0];
    const unsigned char* mask  = (const unsigned char*)d_in[1];
    const float*         W3    = (const float*)d_in[11];
    const float*         b3    = (const float*)d_in[12];
    const float*         We    = (const float*)d_in[13];
    const float*         Wq    = (const float*)d_in[14];
    const float*         battn = (const float*)d_in[15];
    const float*         vvec  = (const float*)d_in[16];
    float*               out   = (float*)d_out;

    char* ws = (char*)d_ws;
    unsigned short* WeT    = (unsigned short*)(ws);                 // 512*512*2 = 524288 B
    float*          cvec   = (float*)(ws + 524288);                 // 2048 B
    float*          scores = (float*)(ws + 526336);                 // 32*2048*4 = 262144 B
    float*          maxv   = (float*)(ws + 788480);                 // 128 B
    float*          sumv   = (float*)(ws + 788608);                 // 128 B

    k_weT   <<<(DD * DD) / 256, 256, 0, stream>>>(We, WeT);
    k_cvec  <<<1, 512, 0, stream>>>(W3, b3, Wq, battn, cvec);
    k_scores<<<(BB * SS) / 64, 256, 0, stream>>>(x, WeT, cvec, vvec, scores);
    k_maxsum<<<BB, 256, 0, stream>>>(mask, scores, maxv, sumv, out);
    k_wsum  <<<BB * 8, 512, 0, stream>>>(x, scores, maxv, sumv, out);
}
